// GraphClassifier_41034117546450
// MI455X (gfx1250) — matrix operand order repaired and checked by reading, not compiled, not run on hardware
//
#include <hip/hip_runtime.h>
#include <math.h>

// ---------------------------------------------------------------------------
// Types for CDNA5 WMMA (wave32): v_wmma_f32_16x16x32_bf16
// ---------------------------------------------------------------------------
typedef __attribute__((ext_vector_type(16))) __bf16      v16bf;
typedef __attribute__((ext_vector_type(8)))  float       v8f;
typedef __attribute__((ext_vector_type(4)))  unsigned    v4u;
typedef __attribute__((ext_vector_type(8)))  unsigned    v8u;
typedef __attribute__((ext_vector_type(4)))  int         v4i;

#define DEV __device__ __forceinline__
#define AS1 __attribute__((address_space(1)))
#define AS3 __attribute__((address_space(3)))

// CDNA5 async global->LDS copy (ASYNCcnt path), guarded so compile never breaks
#if defined(__has_builtin)
#if __has_builtin(__builtin_amdgcn_global_load_async_to_lds_b128)
#define ASYNC_LDS_COPY 1
#endif
#endif

DEV unsigned short f32_to_bf16_bits(float f) {
  unsigned u = __float_as_uint(f);
  u += 0x7FFFu + ((u >> 16) & 1u);          // round-to-nearest-even
  return (unsigned short)(u >> 16);
}
DEV float bf16_bits_to_f32(unsigned short h) {
  return __uint_as_float(((unsigned)h) << 16);
}

// ---------------------------------------------------------------------------
// WMMA GEMM: C[M, NT*16] = A[M,K](bf16,row-major) x Bpacked + bias
// - Entire packed B staged into LDS once per block (<=64KB) via the CDNA5
//   async global->LDS path (ASYNCcnt), read back as ds_load_b128 fragments
//   -> WMMA's B operand waits on DScnt, not LOADcnt.
// - A fragments double-buffered in registers: global b128 loads for k+1 issue
//   before the WMMA chain of k.
// A fragment: lane<16 -> M=lane, K=0..15 contiguous; lane>=16 ->
// M=lane-16, K=16..31. B packed per (ktile,ntile): lane<16 col=lane
// K=0..15 contiguous, lane>=16 col=lane-16 K=16..31.
// ---------------------------------------------------------------------------
template <int NT>
__global__ __launch_bounds__(256) void wmma_gemm(
    const unsigned short* __restrict__ A,   // [M,K] bf16 bits
    const unsigned short* __restrict__ Bp,  // packed fragments (global)
    const float* __restrict__ bias,         // [NT*16] or nullptr
    float* __restrict__ C,                  // [M, NT*16] fp32
    int M, int K) {
  constexpr int N = NT * 16;
  extern __shared__ __align__(16) unsigned short Bs[];

  // ---- stage packed B into LDS (whole matrix, once per block) ----
  const int nB = (K >> 5) * NT * 512;       // elements
#if ASYNC_LDS_COPY
  for (int idx = threadIdx.x * 8; idx < nB; idx += blockDim.x * 8) {
    __builtin_amdgcn_global_load_async_to_lds_b128(
        (AS1 v4i*)(Bp + idx),
        (AS3 v4i*)(Bs + idx), 0, 0);
  }
#if __has_builtin(__builtin_amdgcn_s_wait_asynccnt)
  __builtin_amdgcn_s_wait_asynccnt(0);
#else
  asm volatile("s_wait_asynccnt 0x0" ::: "memory");
#endif
#else
  for (int idx = threadIdx.x * 8; idx < nB; idx += blockDim.x * 8)
    *reinterpret_cast<v4u*>(Bs + idx) = *reinterpret_cast<const v4u*>(Bp + idx);
#endif
  __syncthreads();

  const int lane  = threadIdx.x & 31;
  const int wave  = blockIdx.x * (blockDim.x >> 5) + (threadIdx.x >> 5);
  const int mtiles = M >> 4;                // M is a multiple of 16 here
  if (wave >= mtiles) return;
  const int m0   = wave << 4;
  const int half = lane >> 4;
  const unsigned short* Arow = A + (size_t)(m0 + (lane & 15)) * K;

  v8f acc[NT];
#pragma unroll
  for (int t = 0; t < NT; ++t) {
    v8f z = {0.f, 0.f, 0.f, 0.f, 0.f, 0.f, 0.f, 0.f};
    acc[t] = z;
  }

  // A register double-buffer
  v4u alo = *reinterpret_cast<const v4u*>(Arow + half * 16);
  v4u ahi = *reinterpret_cast<const v4u*>(Arow + 8 + half * 16);

  for (int k0 = 0; k0 < K; k0 += 32) {
    v4u nlo = alo, nhi = ahi;
    const int k1 = k0 + 32;
    if (k1 < K) {
      __builtin_prefetch(Arow + k1 + 32, 0, 3);   // global_prefetch_b8
      nlo = *reinterpret_cast<const v4u*>(Arow + k1 + half * 16);
      nhi = *reinterpret_cast<const v4u*>(Arow + k1 + 8 + half * 16);
    }
    v16bf af = __builtin_bit_cast(v16bf, __builtin_shufflevector(alo, ahi, 0, 1, 2, 3, 4, 5, 6, 7));

    const unsigned short* bbase = Bs + (size_t)(k0 >> 5) * NT * 512 + (size_t)lane * 16;
#pragma unroll
    for (int t = 0; t < NT; ++t) {
      const unsigned short* bp = bbase + (size_t)t * 512;
      v4u blo = *reinterpret_cast<const v4u*>(bp);       // ds_load_b128
      v4u bhi = *reinterpret_cast<const v4u*>(bp + 8);   // ds_load_b128
      v16bf bm = __builtin_bit_cast(v16bf, __builtin_shufflevector(blo, bhi, 0, 1, 2, 3, 4, 5, 6, 7));
      acc[t] = __builtin_amdgcn_wmma_f32_16x16x32_bf16(
          false, af, false, bm, (short)0, acc[t], false, false);
    }
    alo = nlo; ahi = nhi;
  }

  // C/D layout (ISA 7.12.2): VGPR v -> row m0 + v + 8*half, col = t*16 + lane%16
  const int r0 = m0 + half * 8;
#pragma unroll
  for (int t = 0; t < NT; ++t) {
    const int col = t * 16 + (lane & 15);
    const float bv = bias ? bias[col] : 0.0f;
#pragma unroll
    for (int v = 0; v < 8; ++v)
      C[(size_t)(r0 + v) * N + col] = acc[t][v] + bv;
  }
}

// ---------------------------------------------------------------------------
// Weight packing into the B-fragment layout (fp32 -> bf16), with column offset
// so the three inception branches concatenate into one N=160 GEMM.
// ---------------------------------------------------------------------------
__global__ void pack_b(const float* __restrict__ W, unsigned short* __restrict__ dst,
                       int K, int oc, int noff, int Ntot) {
  int idx = blockIdx.x * blockDim.x + threadIdx.x;
  if (idx >= K * oc) return;
  int k = idx / oc, nl = idx - k * oc;
  int n = noff + nl;
  int kt = k >> 5, kk = k & 31;
  int nt = n >> 4, nn = n & 15;
  int lane = (kk < 16) ? nn : nn + 16;
  int v    = (kk < 16) ? kk : kk - 16;
  int NT = Ntot >> 4;
  dst[(((size_t)kt * NT + nt) * 32 + lane) * 16 + v] = f32_to_bf16_bits(W[idx]);
}

// ---------------------------------------------------------------------------
// Elementwise / reduction / graph kernels
// ---------------------------------------------------------------------------
__global__ void fill_f32(float* __restrict__ p, float v, int n) {
  int i = blockIdx.x * blockDim.x + threadIdx.x;
  if (i < n) p[i] = v;
}
__global__ void copy_f32(float* __restrict__ d, const float* __restrict__ s, int n) {
  int i = blockIdx.x * blockDim.x + threadIdx.x;
  if (i < n) d[i] = s[i];
}
__global__ void convert_f32_bf16(const float* __restrict__ s,
                                 unsigned short* __restrict__ d, long long n) {
  long long i = (long long)blockIdx.x * blockDim.x + threadIdx.x;
  if (i < n) d[i] = f32_to_bf16_bits(s[i]);
}
__global__ void deg_accum(const int* __restrict__ dst, float* __restrict__ deg, int E) {
  int e = blockIdx.x * blockDim.x + threadIdx.x;
  if (e < E) atomicAdd(&deg[dst[e]], 1.0f);
}
__global__ void rsqrt_inplace(float* __restrict__ p, int n) {
  int i = blockIdx.x * blockDim.x + threadIdx.x;
  if (i < n) p[i] = rsqrtf(p[i]);
}

// agg[i,:] = t[i,:] * dis[i]^2   (self-loop term: norm = 1/deg)
__global__ void selfloop_init(const float* __restrict__ t, const float* __restrict__ dis,
                              float* __restrict__ agg, int M, int N) {
  long long i = (long long)blockIdx.x * blockDim.x + threadIdx.x;
  if (i >= (long long)M * N) return;
  int r = (int)(i / N);
  float d = dis[r];
  agg[i] = t[i] * d * d;
}

// wave per edge: agg[dst,:] += t[src,:] * dis[src]*dis[dst]  (L2-resident)
__global__ void edge_scatter(const int* __restrict__ src, const int* __restrict__ dst,
                             const float* __restrict__ dis, const float* __restrict__ t,
                             float* __restrict__ agg, int E, int Cch) {
  int wid  = blockIdx.x * (blockDim.x >> 5) + (threadIdx.x >> 5);
  int lane = threadIdx.x & 31;
  if (wid >= E) return;
  int s = src[wid], d = dst[wid];
  float nrm = dis[s] * dis[d];
  const float* ts = t + (size_t)s * Cch;
  float* ad = agg + (size_t)d * Cch;
  for (int c = lane; c < Cch; c += 32) atomicAdd(&ad[c], ts[c] * nrm);
}

// Per-column sum/sumsq (blockDim.x == Ncols, coalesced rows, atomic combine)
__global__ void col_stats(const float* __restrict__ X, float* __restrict__ sum,
                          float* __restrict__ sumsq, int M, int N, int rpb) {
  int c = threadIdx.x;
  int r0 = blockIdx.x * rpb;
  int r1 = min(r0 + rpb, M);
  float s = 0.f, s2 = 0.f;
  for (int r = r0; r < r1; ++r) {
    float v = X[(size_t)r * N + c];
    s += v; s2 += v * v;
  }
  atomicAdd(&sum[c], s);
  atomicAdd(&sumsq[c], s2);
}

// BatchNorm(train, biased var) + ReLU, emit bf16 activations
__global__ void bn_relu_bf16(const float* __restrict__ X, const float* __restrict__ sum,
                             const float* __restrict__ sumsq, const float* __restrict__ g,
                             const float* __restrict__ be,
                             unsigned short* __restrict__ out, int M, int N) {
  long long i = (long long)blockIdx.x * blockDim.x + threadIdx.x;
  if (i >= (long long)M * N) return;
  int c = (int)(i % N);
  float invM = 1.0f / (float)M;
  float m = sum[c] * invM;
  float var = fmaxf(sumsq[c] * invM - m * m, 0.f);
  float sc = rsqrtf(var + 1e-5f) * g[c];
  float y = (X[i] - m) * sc + be[c];
  out[i] = f32_to_bf16_bits(fmaxf(y, 0.f));
}

// scores[i] = ba2 + sum_j relu(G1[i,j]) * Wa2[j]
__global__ void score_kernel(const float* __restrict__ G1, const float* __restrict__ Wa2,
                             const float* __restrict__ ba2, float* __restrict__ scores,
                             int M, int H) {
  int i = blockIdx.x * blockDim.x + threadIdx.x;
  if (i >= M) return;
  float s = ba2[0];
  const float* g = G1 + (size_t)i * H;
  for (int j = 0; j < H; ++j) s += fmaxf(g[j], 0.f) * Wa2[j];
  scores[i] = s;
}

DEV void atomicMaxF(float* addr, float val) {
  if (val >= 0.f) atomicMax((int*)addr, __float_as_int(val));
  else            atomicMin((unsigned int*)addr, __float_as_uint(val));
}
__global__ void seg_max(const float* __restrict__ sc, const int* __restrict__ c2s,
                        float* __restrict__ smax, int M) {
  int i = blockIdx.x * blockDim.x + threadIdx.x;
  if (i < M) atomicMaxF(&smax[c2s[i]], sc[i]);
}
__global__ void seg_exp(const float* __restrict__ sc, const int* __restrict__ c2s,
                        const float* __restrict__ smax, float* __restrict__ ev,
                        float* __restrict__ denom, int M) {
  int i = blockIdx.x * blockDim.x + threadIdx.x;
  if (i >= M) return;
  int s = c2s[i];
  float e = expf(sc[i] - smax[s]);
  ev[i] = e;
  atomicAdd(&denom[s], e);
}
// wave per node: spot_embs[spot,:] += h[i,:] * (e/denom)
__global__ void pool_kernel(const unsigned short* __restrict__ h,
                            const float* __restrict__ ev, const float* __restrict__ denom,
                            const int* __restrict__ c2s, float* __restrict__ se,
                            int M, int N) {
  int wid  = blockIdx.x * (blockDim.x >> 5) + (threadIdx.x >> 5);
  int lane = threadIdx.x & 31;
  if (wid >= M) return;
  int s = c2s[wid];
  float w = ev[wid] / denom[s];
  const unsigned short* hr = h + (size_t)wid * N;
  float* sr = se + (size_t)s * N;
  for (int c = lane; c < N; c += 32)
    atomicAdd(&sr[c], bf16_bits_to_f32(hr[c]) * w);
}
__global__ void final_gemm(const float* __restrict__ se, const float* __restrict__ Wf,
                           const float* __restrict__ bfv, float* __restrict__ out,
                           int S, int K, int C) {
  int idx = blockIdx.x * blockDim.x + threadIdx.x;
  if (idx >= S * C) return;
  int s = idx / C, c = idx - s * C;
  float acc = bfv[c];
  const float* sr = se + (size_t)s * K;
  for (int k = 0; k < K; ++k) acc += sr[k] * Wf[k * C + c];
  out[idx] = acc;
}

// ---------------------------------------------------------------------------
static void launch_gemm(int NT, const unsigned short* A, const unsigned short* Bp,
                        const float* bias, float* C, int M, int K, hipStream_t s) {
  int mtiles = (M + 15) >> 4;
  dim3 grid((mtiles + 7) / 8), block(256);
  size_t shmem = (size_t)K * NT * 32;   // packed B bytes, <= 64KB
  switch (NT) {
    case 4:  wmma_gemm<4><<<grid, block, shmem, s>>>(A, Bp, bias, C, M, K); break;
    case 8:  wmma_gemm<8><<<grid, block, shmem, s>>>(A, Bp, bias, C, M, K); break;
    case 10: wmma_gemm<10><<<grid, block, shmem, s>>>(A, Bp, bias, C, M, K); break;
    default: break;
  }
}

extern "C" void kernel_launch(void* const* d_in, const int* in_sizes, int n_in,
                              void* d_out, int out_size, void* d_ws, size_t ws_size,
                              hipStream_t stream) {
  const float* x   = (const float*)d_in[0];
  const int*   ei  = (const int*)d_in[1];
  const int*   c2s = (const int*)d_in[2];
  const float* W0  = (const float*)d_in[4];
  const float* g1  = (const float*)d_in[6];
  const float* be1 = (const float*)d_in[7];
  const float* Wa1 = (const float*)d_in[32];
  const float* ba1 = (const float*)d_in[33];
  const float* Wa2 = (const float*)d_in[34];
  const float* ba2 = (const float*)d_in[35];
  const float* Wf  = (const float*)d_in[36];
  const float* bfv = (const float*)d_in[37];

  const int H0 = in_sizes[5];                 // 128
  const int IN = in_sizes[4] / H0;            // 256
  const int Nn = in_sizes[0] / IN;            // 100000
  const int E  = in_sizes[1] / 2;             // 1.6M
  const int oc[3] = { in_sizes[9], in_sizes[13], in_sizes[17] };
  const int NI = oc[0] + oc[1] + oc[2];       // 160
  const int HA = in_sizes[33];                // 64
  const int Cc = in_sizes[37];                // 20
  const int S  = out_size / Cc;               // 5000
  const int* esrc = ei;
  const int* edst = ei + E;

  // ---- workspace carve (256B aligned) ----
  char* w = (char*)d_ws;
  size_t off = 0;
  auto alloc = [&](size_t bytes) -> void* {
    void* p = w + off;
    off = (off + bytes + 255) & ~(size_t)255;
    return p;
  };
  unsigned short* xb   = (unsigned short*)alloc((size_t)Nn * IN * 2);
  float*          tbuf = (float*)alloc((size_t)Nn * NI * 4);   // GEMM out / G1
  float*          agg  = (float*)alloc((size_t)Nn * NI * 4);
  unsigned short* hb1  = (unsigned short*)alloc((size_t)Nn * NI * 2);
  unsigned short* hb2  = (unsigned short*)alloc((size_t)Nn * NI * 2);
  float*          dis  = (float*)alloc((size_t)Nn * 4);
  unsigned short* W0p  = (unsigned short*)alloc((size_t)IN * H0 * 2);
  unsigned short* Wc1p = (unsigned short*)alloc((size_t)H0 * NI * 2);
  unsigned short* Wc2p = (unsigned short*)alloc((size_t)NI * NI * 2);
  unsigned short* Wa1p = (unsigned short*)alloc((size_t)NI * HA * 2);
  float* gc1  = (float*)alloc(NI * 4);
  float* bec1 = (float*)alloc(NI * 4);
  float* gc2  = (float*)alloc(NI * 4);
  float* bec2 = (float*)alloc(NI * 4);
  float* csum = (float*)alloc(NI * 4);
  float* csq  = (float*)alloc(NI * 4);
  float* scores = (float*)alloc((size_t)Nn * 4);
  float* evals  = (float*)alloc((size_t)Nn * 4);
  float* smax   = (float*)alloc((size_t)S * 4);
  float* denom  = (float*)alloc((size_t)S * 4);
  float* sembs  = (float*)alloc((size_t)S * NI * 4);

  const int RPB = 512;
  const int statBlocks = (Nn + RPB - 1) / RPB;

  // ---- degrees & symmetric norm (self-loops folded in: deg init = 1) ----
  fill_f32<<<(Nn + 255) / 256, 256, 0, stream>>>(dis, 1.0f, Nn);
  deg_accum<<<(E + 255) / 256, 256, 0, stream>>>(edst, dis, E);
  rsqrt_inplace<<<(Nn + 255) / 256, 256, 0, stream>>>(dis, Nn);

  // ---- convert x to bf16; pack all weights into WMMA B-fragment layout ----
  long long nx = (long long)Nn * IN;
  convert_f32_bf16<<<(unsigned)((nx + 255) / 256), 256, 0, stream>>>(x, xb, nx);
  pack_b<<<(IN * H0 + 255) / 256, 256, 0, stream>>>(W0, W0p, IN, H0, 0, H0);
  {
    int o = 0;
    for (int b = 0; b < 3; ++b) {
      pack_b<<<(H0 * oc[b] + 255) / 256, 256, 0, stream>>>(
          (const float*)d_in[8 + b * 4], Wc1p, H0, oc[b], o, NI);
      copy_f32<<<(oc[b] + 63) / 64, 64, 0, stream>>>(gc1 + o, (const float*)d_in[10 + b * 4], oc[b]);
      copy_f32<<<(oc[b] + 63) / 64, 64, 0, stream>>>(bec1 + o, (const float*)d_in[11 + b * 4], oc[b]);
      o += oc[b];
    }
    o = 0;
    for (int b = 0; b < 3; ++b) {
      pack_b<<<(NI * oc[b] + 255) / 256, 256, 0, stream>>>(
          (const float*)d_in[20 + b * 4], Wc2p, NI, oc[b], o, NI);
      copy_f32<<<(oc[b] + 63) / 64, 64, 0, stream>>>(gc2 + o, (const float*)d_in[22 + b * 4], oc[b]);
      copy_f32<<<(oc[b] + 63) / 64, 64, 0, stream>>>(bec2 + o, (const float*)d_in[23 + b * 4], oc[b]);
      o += oc[b];
    }
  }
  pack_b<<<(NI * HA + 255) / 256, 256, 0, stream>>>(Wa1, Wa1p, NI, HA, 0, HA);

  // ---- layer 1: h = relu(BN(x @ W0))  (b0 absorbed by BN mean-subtract) ----
  launch_gemm(H0 / 16, xb, W0p, nullptr, tbuf, Nn, IN, stream);
  fill_f32<<<1, 256, 0, stream>>>(csum, 0.f, NI);
  fill_f32<<<1, 256, 0, stream>>>(csq, 0.f, NI);
  col_stats<<<statBlocks, H0, 0, stream>>>(tbuf, csum, csq, Nn, H0, RPB);
  {
    long long n = (long long)Nn * H0;
    bn_relu_bf16<<<(unsigned)((n + 255) / 256), 256, 0, stream>>>(
        tbuf, csum, csq, g1, be1, hb1, Nn, H0);
  }

  // ---- two inception blocks (branch GEMMs fused, GCN bias absorbed by BN) ----
  auto inception = [&](const unsigned short* hin, int Kdim, const unsigned short* Wp,
                       const float* gcat, const float* becat, unsigned short* hout) {
    launch_gemm(NI / 16, hin, Wp, nullptr, tbuf, Nn, Kdim, stream);
    long long n = (long long)Nn * NI;
    selfloop_init<<<(unsigned)((n + 255) / 256), 256, 0, stream>>>(tbuf, dis, agg, Nn, NI);
    edge_scatter<<<(E + 7) / 8, 256, 0, stream>>>(esrc, edst, dis, tbuf, agg, E, NI);
    fill_f32<<<1, 256, 0, stream>>>(csum, 0.f, NI);
    fill_f32<<<1, 256, 0, stream>>>(csq, 0.f, NI);
    col_stats<<<statBlocks, NI, 0, stream>>>(agg, csum, csq, Nn, NI, RPB);
    bn_relu_bf16<<<(unsigned)((n + 255) / 256), 256, 0, stream>>>(
        agg, csum, csq, gcat, becat, hout, Nn, NI);
  };
  inception(hb1, H0, Wc1p, gc1, bec1, hb2);
  inception(hb2, NI, Wc2p, gc2, bec2, hb1);   // final h in hb1 [Nn, NI]

  // ---- attention pooling over spots ----
  launch_gemm(HA / 16, hb1, Wa1p, ba1, tbuf, Nn, NI, stream);  // G1 = h@Wa1+ba1
  score_kernel<<<(Nn + 255) / 256, 256, 0, stream>>>(tbuf, Wa2, ba2, scores, Nn, HA);
  fill_f32<<<(S + 255) / 256, 256, 0, stream>>>(smax, -3.402823466e38f, S);
  seg_max<<<(Nn + 255) / 256, 256, 0, stream>>>(scores, c2s, smax, Nn);
  fill_f32<<<(S + 255) / 256, 256, 0, stream>>>(denom, 0.f, S);
  seg_exp<<<(Nn + 255) / 256, 256, 0, stream>>>(scores, c2s, smax, evals, denom, Nn);
  fill_f32<<<(S * NI + 255) / 256, 256, 0, stream>>>(sembs, 0.f, S * NI);
  pool_kernel<<<(Nn + 7) / 8, 256, 0, stream>>>(hb1, evals, denom, c2s, sembs, Nn, NI);

  // ---- classifier head ----
  final_gemm<<<(S * Cc + 255) / 256, 256, 0, stream>>>(sembs, Wf, bfv,
                                                       (float*)d_out, S, NI, Cc);
}
